// TPA_38431367364967
// MI455X (gfx1250) — matrix operand order repaired and checked by reading, not compiled, not run on hardware
//
#include <hip/hip_runtime.h>

// Problem constants (from reference)
#define DIMX 2048
#define NHX  16
#define HDX  128
#define QRX  12
#define RX   2
#define BSZX 2
#define SEQX 2048
#define AKLD 64        // padded leading dim for A_k / A_v projections (real N = 32)

typedef __attribute__((ext_vector_type(16))) __bf16 bf16x16;
typedef __attribute__((ext_vector_type(8)))  float  f32x8;

// ---------------------------------------------------------------------------
// WMMA fragment loaders (contiguous k = 16*(lane/16) + s order, same for A and B, wave32)
// A (16x32 bf16, row-major src): lane m=lane&15 holds row m; lanes>=16 hold
// K-half offset 16. VGPR i holds K = kh + 2i (+1 in hi half).
// Loads collapse to 16 contiguous bf16 per lane.
// ---------------------------------------------------------------------------
__device__ __forceinline__ bf16x16 frag_a_rowmajor(const __bf16* A, int lda, int lane) {
  int m  = lane & 15;
  int kh = (lane >> 4) << 4;                 // 0 or 16
  const __bf16* base = A + (size_t)m * lda + kh;
  bf16x16 f;
#pragma unroll
  for (int i = 0; i < 8; ++i) {
    int k = i << 1;
    f[2 * i]     = base[k];
    f[2 * i + 1] = base[k + 1];
  }
  return f;
}

// B fragment from a TRANSPOSED operand Bt (N x K row-major == original K x N
// column-major): lane n=lane&15 owns column n; lanes>=16 take K-half offset 16.
// VGPR j holds K = kh + 2j, 2j+1  ->  16 contiguous bf16 per lane (2x b128).
__device__ __forceinline__ bf16x16 frag_b_tr(const __bf16* Bt, int ldb, int lane) {
  int n  = lane & 15;
  int kh = (lane >> 4) << 4;                 // 0 or 16
  const __bf16* p = Bt + (size_t)n * ldb + kh;
  bf16x16 f;
#pragma unroll
  for (int j = 0; j < 16; ++j) f[j] = p[j];
  return f;
}

// ---------------------------------------------------------------------------
// fp32 -> bf16 elementwise convert
// ---------------------------------------------------------------------------
__global__ void f32_to_bf16_kernel(const float* __restrict__ s, __bf16* __restrict__ d, size_t n) {
  size_t i      = (size_t)blockIdx.x * blockDim.x + threadIdx.x;
  size_t stride = (size_t)gridDim.x * blockDim.x;
  for (; i < n; i += stride) d[i] = (__bf16)s[i];
}

// ---------------------------------------------------------------------------
// Transpose + convert + pad: src f32 [K x Nsrc] -> dst bf16 [Ndst x K].
// Rows n in [Nsrc, Ndst) are zero-filled so padded GEMM columns are clean.
// ---------------------------------------------------------------------------
__global__ __launch_bounds__(256) void transpose_f32_to_bf16_kernel(
    const float* __restrict__ src, __bf16* __restrict__ dst, int K, int Nsrc, int Ndst) {
  __shared__ float tile[32][33];
  int tx = threadIdx.x & 31;
  int ty = threadIdx.x >> 5;                 // 0..7
#pragma unroll
  for (int i = ty; i < 32; i += 8) {
    int k = blockIdx.y * 32 + i;
    int n = blockIdx.x * 32 + tx;
    tile[i][tx] = (k < K && n < Nsrc) ? src[(size_t)k * Nsrc + n] : 0.f;
  }
  __syncthreads();
#pragma unroll
  for (int i = ty; i < 32; i += 8) {
    int n = blockIdx.x * 32 + i;
    int k = blockIdx.y * 32 + tx;
    if (n < Ndst && k < K) dst[(size_t)n * K + k] = (__bf16)tile[tx][i];
  }
}

// ---------------------------------------------------------------------------
// GEMM: C[M,N] (f32) = A[M,K] (bf16 row-major) @ Bt[N,K] (bf16, B transposed)
// Requires M % 64 == 0, N % 64 == 0, K % 32 == 0 -> branch-free inner loop.
// Wave computes a 32x64 C tile: 2 A-frags x 4 B-frags = 8 WMMA per K-step.
// Block = 8 waves arranged 2(M) x 4(N) -> 64 x 256 of C per block.
// ---------------------------------------------------------------------------
__global__ __launch_bounds__(256) void gemm_bf16_wmma(
    const __bf16* __restrict__ A, const __bf16* __restrict__ Bt,
    float* __restrict__ C, int M, int N, int Kd) {
  int lane  = threadIdx.x & 31;
  int wave  = threadIdx.x >> 5;
  int wm    = wave >> 2;                     // 0..1
  int wn    = wave & 3;                      // 0..3
  int rbase = (blockIdx.y * 2 + wm) * 32;    // first of 32 rows
  int cbase = (blockIdx.x * 4 + wn) * 64;    // first of 64 cols
  if (rbase >= M || cbase >= N) return;      // wave-uniform

  const __bf16* Arow0 = A + (size_t)rbase * Kd;
  const __bf16* Arow1 = A + (size_t)(rbase + 16) * Kd;

  f32x8 acc[2][4];
#pragma unroll
  for (int u = 0; u < 2; ++u)
#pragma unroll
    for (int t = 0; t < 4; ++t)
      acc[u][t] = (f32x8){0.f, 0.f, 0.f, 0.f, 0.f, 0.f, 0.f, 0.f};

#pragma unroll 2
  for (int k0 = 0; k0 < Kd; k0 += 32) {
    __builtin_prefetch(Arow0 + k0 + 512, 0, 0);   // global_prefetch_b8 ahead on A
    bf16x16 a0 = frag_a_rowmajor(Arow0 + k0, Kd, lane);
    bf16x16 a1 = frag_a_rowmajor(Arow1 + k0, Kd, lane);
#pragma unroll
    for (int t = 0; t < 4; ++t) {
      bf16x16 b = frag_b_tr(Bt + (size_t)(cbase + t * 16) * Kd + k0, Kd, lane);
      acc[0][t] = __builtin_amdgcn_wmma_f32_16x16x32_bf16(false, a0, false, b, (short)0, acc[0][t], false, false);
      acc[1][t] = __builtin_amdgcn_wmma_f32_16x16x32_bf16(false, a1, false, b, (short)0, acc[1][t], false, false);
    }
  }

  int moff = ((lane >> 4) << 3);
  int nlow = lane & 15;
#pragma unroll
  for (int u = 0; u < 2; ++u) {
    int m0 = rbase + u * 16 + moff;
#pragma unroll
    for (int t = 0; t < 4; ++t) {
      int n = cbase + t * 16 + nlow;
#pragma unroll
      for (int r = 0; r < 8; ++r)
        C[(size_t)(m0 + r) * N + n] = acc[u][t][r];
    }
  }
}

// ---------------------------------------------------------------------------
// RoPE + rank contraction: builds q,k (b,h,s,d) and v transposed (b,h,d,s), bf16.
// One block per (b,s); thread d = 0..127. A_k/A_v rows have padded stride AKLD.
// ---------------------------------------------------------------------------
__global__ __launch_bounds__(128) void rope_contract_kernel(
    const float* __restrict__ Aq, const float* __restrict__ Ak, const float* __restrict__ Av,
    const float* __restrict__ Bq, const float* __restrict__ Bk, const float* __restrict__ Bv,
    const float* __restrict__ fcos, const float* __restrict__ fsin,
    __bf16* __restrict__ qT, __bf16* __restrict__ kT, __bf16* __restrict__ vT) {
  int b = blockIdx.y, s = blockIdx.x, d = threadIdx.x;
  size_t row = (size_t)b * SEQX + s;

  __shared__ float sAq[NHX * QRX];
  __shared__ float sAk[NHX * RX];
  __shared__ float sAv[NHX * RX];
  for (int i = threadIdx.x; i < NHX * QRX; i += blockDim.x)
    sAq[i] = Aq[row * (NHX * QRX) + i] * (1.0f / QRX);
  if (threadIdx.x < NHX * RX) {
    sAk[threadIdx.x] = Ak[row * AKLD + threadIdx.x] * (1.0f / RX);
    sAv[threadIdx.x] = Av[row * AKLD + threadIdx.x] * (1.0f / RX);
  }
  __syncthreads();

  int  pair = d >> 1;
  bool odd  = d & 1;
  float c  = fcos[(size_t)s * (HDX / 2) + pair];
  float sn = fsin[(size_t)s * (HDX / 2) + pair];

  float bq[QRX];
#pragma unroll
  for (int r = 0; r < QRX; ++r) {
    const float* base = Bq + (row * QRX + r) * HDX;
    float xr = base[2 * pair], xi = base[2 * pair + 1];
    bq[r] = odd ? (xr * sn + xi * c) : (xr * c - xi * sn);
  }
  float bk[RX], bv[RX];
#pragma unroll
  for (int r = 0; r < RX; ++r) {
    const float* baseK = Bk + (row * RX + r) * HDX;
    float xr = baseK[2 * pair], xi = baseK[2 * pair + 1];
    bk[r] = odd ? (xr * sn + xi * c) : (xr * c - xi * sn);
    bv[r] = Bv[(row * RX + r) * HDX + d];
  }

#pragma unroll
  for (int h = 0; h < NHX; ++h) {
    float aq = 0.f;
#pragma unroll
    for (int r = 0; r < QRX; ++r) aq += sAq[h * QRX + r] * bq[r];
    float ak = sAk[h * RX] * bk[0] + sAk[h * RX + 1] * bk[1];
    float av = sAv[h * RX] * bv[0] + sAv[h * RX + 1] * bv[1];

    size_t oq = (((size_t)b * NHX + h) * SEQX + s) * HDX + d;
    qT[oq] = (__bf16)aq;
    kT[oq] = (__bf16)ak;
    vT[(((size_t)b * NHX + h) * HDX + d) * SEQX + s] = (__bf16)av;
  }
}

// ---------------------------------------------------------------------------
// Flash attention, causal. One wave owns 16 query rows; 8 waves/block.
// q,k: (b,h,s,d) bf16; v: (b,h,d,s) bf16; out attn: (b,s,h*128+d) bf16.
// ---------------------------------------------------------------------------
__global__ __launch_bounds__(256) void flash_attn_kernel(
    const __bf16* __restrict__ qg, const __bf16* __restrict__ kg,
    const __bf16* __restrict__ vg, __bf16* __restrict__ attn) {
  __shared__ __bf16 plds[8][16 * 32];        // per-wave P staging (C-layout -> A-layout)

  int lane = threadIdx.x & 31;
  int wave = threadIdx.x >> 5;
  int ln   = lane & 15;
  int hf   = lane >> 4;                      // lane half (0/1)
  int bh   = blockIdx.y;                     // 0..31
  int b    = bh >> 4;
  int h    = bh & 15;
  int q0   = (blockIdx.x * 8 + wave) * 16;   // first query row of this wave

  const __bf16* Qh = qg + (size_t)bh * SEQX * HDX;
  const __bf16* Kh = kg + (size_t)bh * SEQX * HDX;
  const __bf16* Vh = vg + (size_t)bh * HDX * SEQX;

  // q fragments: 16 rows x 128 dims = 4 A-fragments (K=32 each)
  bf16x16 qf[4];
#pragma unroll
  for (int kk = 0; kk < 4; ++kk)
    qf[kk] = frag_a_rowmajor(Qh + (size_t)q0 * HDX + kk * 32, HDX, lane);

  f32x8 o[8];
#pragma unroll
  for (int t = 0; t < 8; ++t) o[t] = (f32x8){0.f, 0.f, 0.f, 0.f, 0.f, 0.f, 0.f, 0.f};
  float mrun[8], lrun[8];
#pragma unroll
  for (int r = 0; r < 8; ++r) { mrun[r] = -1e30f; lrun[r] = 0.f; }

  const float sc = 0.08838834764831845f;     // 1/sqrt(128)
  __bf16* pl = plds[wave];

  for (int kb = 0; kb < q0 + 16; kb += 32) {
    // ---- scores: two 16x16 tiles over keys [kb, kb+32) ----
    f32x8 s0 = (f32x8){0.f, 0.f, 0.f, 0.f, 0.f, 0.f, 0.f, 0.f};
    f32x8 s1 = s0;
#pragma unroll
    for (int kk = 0; kk < 4; ++kk) {
      // K^T fragment: per lane, 16 contiguous bf16 from key row (kb + tile + ln)
      bf16x16 b0, b1;
      const __bf16* p0p = Kh + (size_t)(kb + ln) * HDX + kk * 32 + (hf << 4);
      const __bf16* p1p = Kh + (size_t)(kb + 16 + ln) * HDX + kk * 32 + (hf << 4);
#pragma unroll
      for (int j = 0; j < 16; ++j) { b0[j] = p0p[j]; b1[j] = p1p[j]; }
      s0 = __builtin_amdgcn_wmma_f32_16x16x32_bf16(false, qf[kk], false, b0, (short)0, s0, false, false);
      s1 = __builtin_amdgcn_wmma_f32_16x16x32_bf16(false, qf[kk], false, b1, (short)0, s1, false, false);
    }

    // ---- scale + causal mask + online softmax (row stats via 16-lane xor reduce) ----
    float corrv[8];
#pragma unroll
    for (int r = 0; r < 8; ++r) {
      int row = q0 + (hf << 3) + r;
      int c0  = kb + ln;
      int c1  = c0 + 16;
      float v0 = s0[r] * sc + ((c0 <= row) ? 0.f : -1e9f);
      float v1 = s1[r] * sc + ((c1 <= row) ? 0.f : -1e9f);

      float mt = fmaxf(v0, v1);
      mt = fmaxf(mt, __shfl_xor(mt, 1, 32));
      mt = fmaxf(mt, __shfl_xor(mt, 2, 32));
      mt = fmaxf(mt, __shfl_xor(mt, 4, 32));
      mt = fmaxf(mt, __shfl_xor(mt, 8, 32));

      float mnew = fmaxf(mrun[r], mt);
      float corr = __expf(mrun[r] - mnew);
      float p0   = __expf(v0 - mnew);
      float p1   = __expf(v1 - mnew);
      float ls   = p0 + p1;
      ls += __shfl_xor(ls, 1, 32);
      ls += __shfl_xor(ls, 2, 32);
      ls += __shfl_xor(ls, 4, 32);
      ls += __shfl_xor(ls, 8, 32);

      lrun[r]  = lrun[r] * corr + ls;
      mrun[r]  = mnew;
      corrv[r] = corr;

      int rr = (hf << 3) + r;
      pl[rr * 32 + ln]      = (__bf16)p0;
      pl[rr * 32 + 16 + ln] = (__bf16)p1;
    }
    asm volatile("s_wait_dscnt 0" ::: "memory");

#pragma unroll
    for (int t = 0; t < 8; ++t)
#pragma unroll
      for (int r = 0; r < 8; ++r) o[t][r] *= corrv[r];

    // ---- P (A-layout from LDS) @ V(32 keys x 128 dims) ----
    bf16x16 pf;
    {
      int m   = ln;
      int kh8 = hf << 4;
#pragma unroll
      for (int i = 0; i < 8; ++i) {
        int k = kh8 + (i << 1);
        pf[2 * i]     = pl[m * 32 + k];
        pf[2 * i + 1] = pl[m * 32 + k + 1];
      }
    }
#pragma unroll
    for (int t = 0; t < 8; ++t) {
      bf16x16 vf;
      const __bf16* vp = Vh + (size_t)((t << 4) + ln) * SEQX + kb + (hf << 4);
#pragma unroll
      for (int j = 0; j < 16; ++j) vf[j] = vp[j];
      o[t] = __builtin_amdgcn_wmma_f32_16x16x32_bf16(false, pf, false, vf, (short)0, o[t], false, false);
    }
  }

  // ---- epilogue: normalize and store to (b, s, h*128 + d) bf16 ----
#pragma unroll
  for (int r = 0; r < 8; ++r) {
    int row = q0 + (hf << 3) + r;
    float inv = 1.0f / lrun[r];
#pragma unroll
    for (int t = 0; t < 8; ++t) {
      size_t oidx = ((size_t)b * SEQX + row) * (NHX * HDX) + h * HDX + (t << 4) + ln;
      attn[oidx] = (__bf16)(o[t][r] * inv);
    }
  }
}

// ---------------------------------------------------------------------------
// Host orchestration
// ---------------------------------------------------------------------------
extern "C" void kernel_launch(void* const* d_in, const int* in_sizes, int n_in,
                              void* d_out, int out_size, void* d_ws, size_t ws_size,
                              hipStream_t stream) {
  (void)in_sizes; (void)n_in; (void)out_size; (void)ws_size;

  const float* Q   = (const float*)d_in[0];
  const float* Kin = (const float*)d_in[1];
  const float* V   = (const float*)d_in[2];
  const float* fc  = (const float*)d_in[4];
  const float* fs  = (const float*)d_in[5];
  const float* WAq = (const float*)d_in[7];
  const float* WAk = (const float*)d_in[8];
  const float* WAv = (const float*)d_in[9];
  const float* WBq = (const float*)d_in[10];
  const float* WBk = (const float*)d_in[11];
  const float* WBv = (const float*)d_in[12];
  const float* Wo  = (const float*)d_in[13];
  float* out = (float*)d_out;

  const size_t TOK = (size_t)BSZX * SEQX;    // 4096 token rows

  char* ws = (char*)d_ws;
  size_t off = 0;
  auto salloc = [&](size_t elems, size_t esz) -> void* {
    void* p = ws + off;
    off += (elems * esz + 255) & ~(size_t)255;
    return p;
  };

  __bf16* Qb   = (__bf16*)salloc(TOK * DIMX, 2);
  __bf16* Kb   = (__bf16*)salloc(TOK * DIMX, 2);
  __bf16* Vb   = (__bf16*)salloc(TOK * DIMX, 2);
  __bf16* WAqT = (__bf16*)salloc((size_t)DIMX * NHX * QRX, 2);   // 192 x 2048
  __bf16* WAkT = (__bf16*)salloc((size_t)DIMX * AKLD, 2);        // 64 x 2048 (padded)
  __bf16* WAvT = (__bf16*)salloc((size_t)DIMX * AKLD, 2);
  __bf16* WBqT = (__bf16*)salloc((size_t)DIMX * QRX * HDX, 2);   // 1536 x 2048
  __bf16* WBkT = (__bf16*)salloc((size_t)DIMX * RX * HDX, 2);    // 256 x 2048
  __bf16* WBvT = (__bf16*)salloc((size_t)DIMX * RX * HDX, 2);
  __bf16* WoT  = (__bf16*)salloc((size_t)DIMX * DIMX, 2);        // 2048 x 2048
  float*  Aq   = (float*)salloc(TOK * NHX * QRX, 4);
  float*  Ak   = (float*)salloc(TOK * AKLD, 4);                  // padded stride 64
  float*  Av   = (float*)salloc(TOK * AKLD, 4);
  float*  Bq   = (float*)salloc(TOK * QRX * HDX, 4);
  float*  Bk   = (float*)salloc(TOK * RX * HDX, 4);
  float*  Bv   = (float*)salloc(TOK * RX * HDX, 4);
  __bf16* qT   = (__bf16*)salloc(TOK * NHX * HDX, 2);
  __bf16* kT   = (__bf16*)salloc(TOK * NHX * HDX, 2);
  __bf16* vT   = (__bf16*)salloc(TOK * NHX * HDX, 2);
  __bf16* attn = (__bf16*)salloc(TOK * NHX * HDX, 2);

  auto cvt = [&](const float* s, __bf16* d, size_t n) {
    int blocks = (int)((n + 256 * 8 - 1) / (256 * 8));
    if (blocks > 4096) blocks = 4096;
    if (blocks < 1) blocks = 1;
    f32_to_bf16_kernel<<<blocks, 256, 0, stream>>>(s, d, n);
  };
  cvt(Q,   Qb, TOK * DIMX);
  cvt(Kin, Kb, TOK * DIMX);
  cvt(V,   Vb, TOK * DIMX);

  auto tr = [&](const float* s, __bf16* d, int K, int Nsrc, int Ndst) {
    transpose_f32_to_bf16_kernel<<<dim3((Ndst + 31) / 32, (K + 31) / 32), 256, 0, stream>>>(
        s, d, K, Nsrc, Ndst);
  };
  tr(WAq, WAqT, DIMX, NHX * QRX, NHX * QRX);
  tr(WAk, WAkT, DIMX, NHX * RX,  AKLD);      // pad 32 -> 64 with zeros
  tr(WAv, WAvT, DIMX, NHX * RX,  AKLD);
  tr(WBq, WBqT, DIMX, QRX * HDX, QRX * HDX);
  tr(WBk, WBkT, DIMX, RX * HDX,  RX * HDX);
  tr(WBv, WBvT, DIMX, RX * HDX,  RX * HDX);
  tr(Wo,  WoT,  DIMX, DIMX,      DIMX);

  auto gemm = [&](const __bf16* A, const __bf16* Bt, float* C, int M, int N, int Kd) {
    dim3 g((N + 255) / 256, (M + 63) / 64);
    gemm_bf16_wmma<<<g, 256, 0, stream>>>(A, Bt, C, M, N, Kd);
  };
  gemm(Qb, WAqT, Aq, (int)TOK, NHX * QRX, DIMX);   // 4096 x 192
  gemm(Qb, WBqT, Bq, (int)TOK, QRX * HDX, DIMX);   // 4096 x 1536
  gemm(Kb, WAkT, Ak, (int)TOK, AKLD,      DIMX);   // 4096 x 64 (padded)
  gemm(Kb, WBkT, Bk, (int)TOK, RX * HDX,  DIMX);   // 4096 x 256
  gemm(Vb, WAvT, Av, (int)TOK, AKLD,      DIMX);   // 4096 x 64 (padded)
  gemm(Vb, WBvT, Bv, (int)TOK, RX * HDX,  DIMX);   // 4096 x 256

  rope_contract_kernel<<<dim3(SEQX, BSZX), 128, 0, stream>>>(
      Aq, Ak, Av, Bq, Bk, Bv, fc, fs, qT, kT, vT);

  flash_attn_kernel<<<dim3(SEQX / 128, BSZX * NHX), 256, 0, stream>>>(qT, kT, vT, attn);

  gemm(attn, WoT, out, (int)TOK, DIMX, DIMX);      // 4096 x 2048 final projection
}
